// MultiHeadAttention_84902913507682
// MI455X (gfx1250) — matrix operand order repaired and checked by reading, not compiled, not run on hardware
//
#include <hip/hip_runtime.h>

#define Hh   16
#define DKc  64
#define DXc  1024
#define Bc   2
#define Lc   2048
#define Mtot (Bc * Lc)     // 4096
#define HDK  (Hh * DKc)    // 1024

typedef __bf16 bf16x16 __attribute__((ext_vector_type(16)));
typedef float  f32x8   __attribute__((ext_vector_type(8)));
typedef unsigned int u32x4 __attribute__((ext_vector_type(4)));

union Frag16 { u32x4 q[2]; bf16x16 v; };

__device__ __forceinline__ unsigned short f2bf(float f) {
  unsigned int u = __float_as_uint(f);
  unsigned int r = u + 0x7FFFu + ((u >> 16) & 1u);   // round-to-nearest-even
  return (unsigned short)(r >> 16);
}

// A-matrix fragment (16x32 bf16): lanes 0-15 hold K=0..7 & 16..23 of row M=lane;
// lanes 16-31 hold K=8..15 & 24..31 of row M=lane-16.
__device__ __forceinline__ bf16x16 load_frag_a(const unsigned short* rowp, int lane) {
  int o = (lane < 16) ? 0 : 8;
  Frag16 f;
  f.q[0] = *(const u32x4*)(rowp + o);
  f.q[1] = *(const u32x4*)(rowp + o + 16);
  return f.v;
}

// B-matrix fragment (32x16 bf16): lanes 0-15 hold K=0..7 & 16..23 of col N=lane;
// lanes 16-31 hold K=8..15 & 24..31 of col N=lane-16.
__device__ __forceinline__ bf16x16 load_frag_b(const unsigned short* colp, int lane) {
  int o = (lane < 16) ? 0 : 8;
  Frag16 f;
  f.q[0] = *(const u32x4*)(colp + o);
  f.q[1] = *(const u32x4*)(colp + o + 16);
  return f.v;
}

__device__ __forceinline__ f32x8 wmma_bf(bf16x16 a, bf16x16 b, f32x8 c) {
  return __builtin_amdgcn_wmma_f32_16x16x32_bf16(false, a, false, b, (short)0, c,
                                                 false, false);
}

// Async copy of one 16B chunk from global memory into LDS (per-lane), ASYNCcnt.
__device__ __forceinline__ void async_cp16(unsigned lds_off, unsigned long long gaddr) {
  asm volatile("global_load_async_to_lds_b128 %0, %1, off"
               :: "v"(lds_off), "v"(gaddr) : "memory");
}
__device__ __forceinline__ void wait_async0() {
  asm volatile("s_wait_asynccnt 0x0" ::: "memory");
}

// ---------------------------------------------------------------- pack kernels
__global__ void pack_bf16_kernel(const float* __restrict__ src,
                                 unsigned short* __restrict__ dst, int n) {
  int i = blockIdx.x * blockDim.x + threadIdx.x;
  int stride = gridDim.x * blockDim.x;
  for (; i < n; i += stride) dst[i] = f2bf(src[i]);
}

// dst[n][k] = bf16(src[k][n])  (transpose so WMMA B-fragments read contiguously)
__global__ void pack_wT_kernel(const float* __restrict__ src,
                               unsigned short* __restrict__ dst, int K, int N) {
  int i = blockIdx.x * blockDim.x + threadIdx.x;
  int total = K * N;
  int stride = gridDim.x * blockDim.x;
  for (; i < total; i += stride) {
    int k = i / N, n = i - k * N;
    dst[(size_t)n * K + k] = f2bf(src[i]);
  }
}

// ---------------------------------------------------------------- GEMM kernels
// C[M=4096, N=1024] = A(bf16) * Wt(bf16,[N][K]) + bias
// MODE 0: store bf16 into [b,h,l,d]   (Q, K projections)
// MODE 1: store bf16 into [b,h,d,l]   (V projection, transposed)
// MODE 2: store f32  row-major + residual (output projection)
template <int MODE>
__global__ void __launch_bounds__(128)
gemm_bf16_kernel(const unsigned short* __restrict__ A,
                 const unsigned short* __restrict__ Wt,
                 const float* __restrict__ bias,
                 const float* __restrict__ resid,
                 unsigned short* __restrict__ outB,
                 float* __restrict__ outF) {
  const int lane = threadIdx.x & 31;
  const int w    = threadIdx.x >> 5;
  const int l16  = lane & 15;
  const int hi8  = (lane >> 4) << 3;
  const int mbase = blockIdx.x * 64 + w * 16;
  const int nbase = blockIdx.y * 64;

  const unsigned short* Arow = A + (size_t)(mbase + l16) * DXc;

  f32x8 zero = {};
  f32x8 acc[4] = {zero, zero, zero, zero};

  for (int kb = 0; kb < DXc; kb += 32) {
    __builtin_prefetch(Arow + kb + 128, 0, 1);
    bf16x16 af = load_frag_a(Arow + kb, lane);
#pragma unroll
    for (int t = 0; t < 4; ++t) {
      const unsigned short* Wrow = Wt + (size_t)(nbase + t * 16 + l16) * DXc + kb;
      bf16x16 wf = load_frag_b(Wrow, lane);
      acc[t] = wmma_bf(af, wf, acc[t]);
    }
  }

#pragma unroll
  for (int t = 0; t < 4; ++t) {
#pragma unroll
    for (int i = 0; i < 8; ++i) {
      int row = mbase + i + hi8;          // 0..4095 = b*Lc + l
      int col = nbase + t * 16 + l16;     // 0..1023 = h*DKc + d
      float val = acc[t][i] + bias[col];
      if (MODE == 2) {
        size_t idx = (size_t)row * DXc + col;
        outF[idx] = val + resid[idx];
      } else {
        int bb = row >> 11, l = row & (Lc - 1);
        int hh = col >> 6,  d = col & (DKc - 1);
        if (MODE == 0)
          outB[(((size_t)(bb * Hh + hh)) * Lc + l) * DKc + d] = f2bf(val);
        else
          outB[(((size_t)(bb * Hh + hh)) * DKc + d) * Lc + l] = f2bf(val);
      }
    }
  }
}

// ---------------------------------------------------------------- attention
// grid (B*H, Lq/64), block 128 (4 waves). Wave handles 16 q-rows x full Lk.
// K tiles (pass 1) and V tiles (pass 2) are shared by all 4 waves, so they are
// staged in LDS via double-buffered GLOBAL_LOAD_ASYNC_TO_LDS_B128.
__global__ void __launch_bounds__(128)
attn_kernel(const unsigned short* __restrict__ Qb,
            const unsigned short* __restrict__ Kb,
            const unsigned short* __restrict__ Vt,
            const int* __restrict__ mask,
            float* __restrict__ attnp,
            unsigned short* __restrict__ Ob) {
  const int lane = threadIdx.x & 31;
  const int w    = threadIdx.x >> 5;
  const int l16  = lane & 15;
  const int hi8  = (lane >> 4) << 3;
  const int bh   = blockIdx.x;              // b*H + h
  const int b    = bh >> 4;
  const int h    = bh & 15;
  const int maskb = bh % Bc;                // reference mask-tiling quirk
  const int qbase = blockIdx.y * 64 + w * 16;

  __shared__ __align__(16) unsigned short kst[2][16 * DKc];   // 2 x 2KB K tile
  __shared__ __align__(16) unsigned short vst[2][DKc * 32];   // 2 x 4KB V tile
  __shared__ __align__(16) unsigned short pt[4][16][48];      // P transpose (96B pitch)

  const unsigned short* Qrow = Qb + ((size_t)bh * Lc + qbase + l16) * DKc;
  bf16x16 qa0 = load_frag_a(Qrow, lane);
  bf16x16 qa1 = load_frag_a(Qrow + 32, lane);

  // async-stage one 16-row K tile (16 x 64 bf16 = 2KB contiguous): 16B/thread
  auto issueK = [&](int nt, int bsel) {
    const unsigned short* src = Kb + ((size_t)bh * Lc + nt * 16) * DKc;
    unsigned dst = (unsigned)(uintptr_t)&kst[bsel][0];
    int t = threadIdx.x;
    async_cp16(dst + t * 16, (unsigned long long)(uintptr_t)(src + t * 8));
  };
  // async-stage one V tile (64 d-rows x 32 kpos = 4KB): 32B/thread (2 chunks)
  auto issueV = [&](int kc, int bsel) {
    int t = threadIdx.x;
    int r = t >> 1, half = t & 1;
    const unsigned short* src = Vt + ((size_t)bh * DKc + r) * Lc + kc + half * 16;
    unsigned dst = (unsigned)(uintptr_t)&vst[bsel][r * 32 + half * 16];
    async_cp16(dst,      (unsigned long long)(uintptr_t)src);
    async_cp16(dst + 16, (unsigned long long)(uintptr_t)(src + 8));
  };

  // per-lane online softmax state (merged across lanes once at the end)
  float lm[8], ls[8];
#pragma unroll
  for (int i = 0; i < 8; ++i) { lm[i] = -3.0e38f; ls[i] = 0.0f; }

  // ---- pass 1: scores + per-lane online softmax statistics ----
  issueK(0, 0);
  for (int nt = 0; nt < Lc / 16; ++nt) {
    wait_async0();
    __syncthreads();
    if (nt + 1 < Lc / 16) issueK(nt + 1, (nt + 1) & 1);

    const unsigned short* kr = &kst[nt & 1][l16 * DKc];
    bf16x16 kb0 = load_frag_b(kr, lane);
    bf16x16 kb1 = load_frag_b(kr + 32, lane);
    f32x8 acc = {};
    acc = wmma_bf(qa0, kb0, acc);
    acc = wmma_bf(qa1, kb1, acc);

    int col = nt * 16 + l16;
#pragma unroll
    for (int i = 0; i < 8; ++i) {
      float sc = acc[i] * 0.125f;         // / sqrt(DK)
      int row = qbase + i + hi8;
      if (mask[((size_t)maskb * Lc + row) * Lc + col] != 0) sc = -1.0e9f;
      attnp[((size_t)bh * Lc + row) * Lc + col] = sc;   // raw score (scratch)
      float nm = fmaxf(lm[i], sc);
      ls[i] = ls[i] * __expf(lm[i] - nm) + __expf(sc - nm);
      lm[i] = nm;
    }
  }

  // merge (m, s) pairs across the 16 lanes of each half (xor masks stay in-half)
  float rm[8], inv[8];
#pragma unroll
  for (int i = 0; i < 8; ++i) {
    float m = lm[i], s = ls[i];
#pragma unroll
    for (int d = 1; d < 16; d <<= 1) {
      float om = __shfl_xor(m, d);
      float os = __shfl_xor(s, d);
      float nm = fmaxf(m, om);
      s = s * __expf(m - nm) + os * __expf(om - nm);
      m = nm;
    }
    rm[i] = m;
    inv[i] = 1.0f / s;
  }

  // ---- pass 2: finalize probs, P*V via WMMA (transpose P through LDS) ----
  f32x8 zero = {};
  f32x8 oacc[4] = {zero, zero, zero, zero};

  issueV(0, 0);
  for (int kci = 0; kci < Lc / 32; ++kci) {
    int kc = kci * 32;
#pragma unroll
    for (int j = 0; j < 2; ++j) {
      int col = kc + j * 16 + l16;
#pragma unroll
      for (int i = 0; i < 8; ++i) {
        int rl = i + hi8;
        size_t idx = ((size_t)bh * Lc + qbase + rl) * Lc + col;
        float p = __expf(attnp[idx] - rm[i]) * inv[i];
        attnp[idx] = p;                               // final probability out
        pt[w][rl][j * 16 + l16] = f2bf(p);            // stage for transpose
      }
    }
    wait_async0();
    __syncthreads();                                  // V tile ready, pt visible
    if (kci + 1 < Lc / 32) issueV(kc + 32, (kci + 1) & 1);

    bf16x16 pa = load_frag_a(&pt[w][l16][0], lane);
#pragma unroll
    for (int t = 0; t < 4; ++t) {
      const unsigned short* vr = &vst[kci & 1][(t * 16 + l16) * 32];
      bf16x16 vb = load_frag_b(vr, lane);
      oacc[t] = wmma_bf(pa, vb, oacc[t]);
    }
  }

#pragma unroll
  for (int t = 0; t < 4; ++t)
#pragma unroll
    for (int i = 0; i < 8; ++i) {
      int row = qbase + i + hi8;
      int col = h * DKc + t * 16 + l16;
      Ob[((size_t)(b * Lc + row)) * HDK + col] = f2bf(oacc[t][i]);
    }
}

// ---------------------------------------------------------------- layer norm
__global__ void __launch_bounds__(256)
ln_kernel(const float* __restrict__ Y, const float* __restrict__ g,
          const float* __restrict__ bta, float* __restrict__ out) {
  const int lane = threadIdx.x & 31;
  const int w    = threadIdx.x >> 5;
  const int row  = blockIdx.x * 8 + w;
  const float* yr = Y + (size_t)row * DXc;
  float s = 0.0f, ss = 0.0f;
  float vals[32];
#pragma unroll
  for (int j = 0; j < 32; ++j) {
    float v = yr[lane + j * 32];
    vals[j] = v;
    s += v;
    ss += v * v;
  }
#pragma unroll
  for (int m = 1; m < 32; m <<= 1) {
    s  += __shfl_xor(s, m);
    ss += __shfl_xor(ss, m);
  }
  float mu  = s * (1.0f / DXc);
  float var = ss * (1.0f / DXc) - mu * mu;
  float r   = rsqrtf(var + 1e-5f);
  float* orow = out + (size_t)row * DXc;
#pragma unroll
  for (int j = 0; j < 32; ++j) {
    int c = lane + j * 32;
    orow[c] = (vals[j] - mu) * r * g[c] + bta[c];
  }
}

// ---------------------------------------------------------------- launcher
extern "C" void kernel_launch(void* const* d_in, const int* in_sizes, int n_in,
                              void* d_out, int out_size, void* d_ws, size_t ws_size,
                              hipStream_t stream) {
  (void)in_sizes; (void)n_in; (void)out_size; (void)ws_size;

  const float* q    = (const float*)d_in[0];
  const float* k    = (const float*)d_in[1];
  const float* v    = (const float*)d_in[2];
  const int*   mask = (const int*)d_in[3];
  const float* wq_w = (const float*)d_in[4];
  const float* wq_b = (const float*)d_in[5];
  const float* wk_w = (const float*)d_in[6];
  const float* wk_b = (const float*)d_in[7];
  const float* wv_w = (const float*)d_in[8];
  const float* wv_b = (const float*)d_in[9];
  const float* wo_w = (const float*)d_in[10];
  const float* wo_b = (const float*)d_in[11];
  const float* ln_g = (const float*)d_in[12];
  const float* ln_b = (const float*)d_in[13];

  float* out   = (float*)d_out;
  float* attnp = out + (size_t)Bc * Lc * DXc;   // attn output region

  char* ws = (char*)d_ws;
  size_t off = 0;
  auto alloc = [&](size_t bytes) -> void* {
    void* p = ws + off;
    off += (bytes + 255) & ~(size_t)255;
    return p;
  };
  unsigned short* xq  = (unsigned short*)alloc((size_t)Mtot * DXc * 2);
  unsigned short* xk  = (unsigned short*)alloc((size_t)Mtot * DXc * 2);
  unsigned short* xv  = (unsigned short*)alloc((size_t)Mtot * DXc * 2);
  unsigned short* wqT = (unsigned short*)alloc((size_t)DXc * HDK * 2);
  unsigned short* wkT = (unsigned short*)alloc((size_t)DXc * HDK * 2);
  unsigned short* wvT = (unsigned short*)alloc((size_t)DXc * HDK * 2);
  unsigned short* woT = (unsigned short*)alloc((size_t)HDK * DXc * 2);
  unsigned short* Qb  = (unsigned short*)alloc((size_t)Mtot * DKc * Hh * 2);
  unsigned short* Kb  = (unsigned short*)alloc((size_t)Mtot * DKc * Hh * 2);
  unsigned short* Vt  = (unsigned short*)alloc((size_t)Mtot * DKc * Hh * 2);
  // reuse: O aliases xq (dead after Q proj); Y aliases xk+xv (dead after K/V proj)
  unsigned short* Ob = xq;
  float*          Yb = (float*)xk;

  const int n_act = Mtot * DXc;   // 4,194,304

  pack_bf16_kernel<<<2048, 256, 0, stream>>>(q, xq, n_act);
  pack_bf16_kernel<<<2048, 256, 0, stream>>>(k, xk, n_act);
  pack_bf16_kernel<<<2048, 256, 0, stream>>>(v, xv, n_act);
  pack_wT_kernel<<<2048, 256, 0, stream>>>(wq_w, wqT, DXc, HDK);
  pack_wT_kernel<<<2048, 256, 0, stream>>>(wk_w, wkT, DXc, HDK);
  pack_wT_kernel<<<2048, 256, 0, stream>>>(wv_w, wvT, DXc, HDK);
  pack_wT_kernel<<<2048, 256, 0, stream>>>(wo_w, woT, HDK, DXc);

  dim3 gg(Mtot / 64, DXc / 64);   // (64, 16)
  gemm_bf16_kernel<0><<<gg, 128, 0, stream>>>(xq, wqT, wq_b, nullptr, Qb, nullptr);
  gemm_bf16_kernel<0><<<gg, 128, 0, stream>>>(xk, wkT, wk_b, nullptr, Kb, nullptr);
  gemm_bf16_kernel<1><<<gg, 128, 0, stream>>>(xv, wvT, wv_b, nullptr, Vt, nullptr);

  attn_kernel<<<dim3(Bc * Hh, Lc / 64), 128, 0, stream>>>(Qb, Kb, Vt, mask, attnp, Ob);

  gemm_bf16_kernel<2><<<gg, 128, 0, stream>>>(Ob, woT, wo_b, q, nullptr, Yb);

  ln_kernel<<<Mtot / 8, 256, 0, stream>>>(Yb, ln_g, ln_b, out);
}
